// MultiHeadSelfAttention_50766513439024
// MI455X (gfx1250) — matrix operand order repaired and checked by reading, not compiled, not run on hardware
//
#include <hip/hip_runtime.h>
#include <hip/hip_bf16.h>

// Problem constants (fixed by the reference).
constexpr int BB  = 2;     // batch
constexpr int TT  = 2048;  // sequence length
constexpr int DD  = 1024;  // model dim
constexpr int HH  = 16;    // heads
constexpr int DKK = 64;    // head dim

typedef __attribute__((ext_vector_type(16))) _Float16 v16h;
typedef __attribute__((ext_vector_type(8)))  _Float16 v8h;
typedef __attribute__((ext_vector_type(8)))  float    v8f;
typedef __attribute__((ext_vector_type(4)))  float    v4f;

// ---------------------------------------------------------------------------
// CDNA5 async global->LDS copy (tracked by ASYNCcnt). 16 bytes per lane.
// VDST operand = LDS byte address (low 32 bits of the flat shared-aperture
// address, per the ISA flat->LDS truncation rule).
// ---------------------------------------------------------------------------
__device__ __forceinline__ void async_copy_b128(const _Float16* lds_dst,
                                                const _Float16* gsrc) {
    unsigned dst = (unsigned)(unsigned long long)lds_dst;
    asm volatile("global_load_async_to_lds_b128 %0, %1, off"
                 :: "v"(dst), "v"(gsrc)
                 : "memory");
}
__device__ __forceinline__ void wait_async0() {
    asm volatile("s_wait_asynccnt 0x0" ::: "memory");
}

// ---------------------------------------------------------------------------
// WMMA fragment loaders from LDS rows (row-major, K contiguous).
// A 16x32 f16: lane l -> M=l%16, g=l/16; halves {0..7}=K g*8.., {8..15}=K 16+g*8..
// B 32x16 f16: lane l -> N=l%16, g=l/16; halves {0..7}=K g*8.., {8..15}=K 16+g*8..
// ---------------------------------------------------------------------------
__device__ inline v16h load_a_frag(const _Float16* row, int g) {
    v8h lo = *(const v8h*)(row + g * 8);
    v8h hi = *(const v8h*)(row + 16 + g * 8);
    return __builtin_shufflevector(lo, hi, 0,1,2,3,4,5,6,7,8,9,10,11,12,13,14,15);
}
__device__ inline v16h load_b_frag(const _Float16* row, int g) {
    v8h lo = *(const v8h*)(row + g * 8);
    v8h hi = *(const v8h*)(row + 16 + g * 8);
    return __builtin_shufflevector(lo, hi, 0,1,2,3,4,5,6,7,8,9,10,11,12,13,14,15);
}

// ---------------------------------------------------------------------------
// Projection GEMM, block tile 64x64, K-step 32, 128 threads = 4 waves (2x2).
// OUT_MODE 0: C = X@W, f32 row-major (ldc).                [normal orient]
// OUT_MODE 1: (X@W)^T orientation, f16 head-split t-major Ch[b][h][t][dk]
//             -> per-lane r direction = dk (contiguous), packed v8h stores.
// OUT_MODE 3: C = X@W, f16 head-split dk-major Ch[b][h][dk][t]
//             -> per-lane r direction = t (contiguous), packed v8h stores.
// Ag = X (tokens x K, lda), Bg = W (K x features, ldb).
// Mode 1 grid: (tokens/64, features/64); modes 0,3: (features/64, tokens/64).
// ---------------------------------------------------------------------------
template <int OUT_MODE>
__global__ __launch_bounds__(128) void proj_gemm_kernel(
    const float* __restrict__ Ag, const float* __restrict__ Bg,
    float* __restrict__ Cg, _Float16* __restrict__ Ch,
    int Kdim, int lda, int ldb, int ldc) {

    constexpr bool TR = (OUT_MODE == 1);

    __shared__ alignas(16) _Float16 As[64][32];
    __shared__ alignas(16) _Float16 Bs[64][32];

    const int tid  = threadIdx.x;
    const int lane = tid & 31;
    const int wave = tid >> 5;
    const int wm   = wave >> 1, wn = wave & 1;
    const int g    = lane >> 4, lr = lane & 15;
    const int m0 = blockIdx.y * 64, n0 = blockIdx.x * 64;

    v8f acc[2][2] = {};

    for (int k0 = 0; k0 < Kdim; k0 += 32) {
        if (TR) {
            #pragma unroll
            for (int i = 0; i < 16; ++i) {
                int idx = i * 128 + tid;
                int c = idx >> 6, f = idx & 63;
                As[f][c] = (_Float16)Bg[(size_t)(k0 + c) * ldb + (m0 + f)];
            }
            #pragma unroll
            for (int i = 0; i < 16; ++i) {
                int idx = i * 128 + tid;
                int n = idx >> 5, c = idx & 31;
                Bs[n][c] = (_Float16)Ag[(size_t)(n0 + n) * lda + (k0 + c)];
            }
        } else {
            #pragma unroll
            for (int i = 0; i < 16; ++i) {
                int idx = i * 128 + tid;
                int r = idx >> 5, c = idx & 31;
                As[r][c] = (_Float16)Ag[(size_t)(m0 + r) * lda + (k0 + c)];
            }
            #pragma unroll
            for (int i = 0; i < 16; ++i) {
                int idx = i * 128 + tid;
                int k = idx >> 6, n = idx & 63;
                Bs[n][k] = (_Float16)Bg[(size_t)(k0 + k) * ldb + (n0 + n)];
            }
        }
        __syncthreads();

        v16h af[2], bf[2];
        #pragma unroll
        for (int fm = 0; fm < 2; ++fm) af[fm] = load_a_frag(&As[wm*32 + fm*16 + lr][0], g);
        #pragma unroll
        for (int fn = 0; fn < 2; ++fn) bf[fn] = load_b_frag(&Bs[wn*32 + fn*16 + lr][0], g);
        #pragma unroll
        for (int fm = 0; fm < 2; ++fm)
            #pragma unroll
            for (int fn = 0; fn < 2; ++fn)
                acc[fm][fn] = __builtin_amdgcn_wmma_f32_16x16x32_f16(
                    false, af[fm], false, bf[fn], (short)0, acc[fm][fn], false, false);
        __syncthreads();
    }

    #pragma unroll
    for (int fm = 0; fm < 2; ++fm)
        #pragma unroll
        for (int fn = 0; fn < 2; ++fn) {
            int ml = m0 + wm * 32 + fm * 16 + g * 8;   // M-base (+r)
            int gn = n0 + wn * 32 + fn * 16 + lr;      // N (per lane)
            if (OUT_MODE == 1) {
                int b = gn >> 11, t = gn & (TT - 1);
                int h = ml >> 6,  dkb = ml & (DKK - 1);
                v8h ph;
                #pragma unroll
                for (int r = 0; r < 8; ++r) ph[r] = (_Float16)acc[fm][fn][r];
                *(v8h*)&Ch[(((size_t)b * HH + h) * TT + t) * DKK + dkb] = ph;
            } else if (OUT_MODE == 3) {
                int b = ml >> 11, tb = ml & (TT - 1);
                int h = gn >> 6,  dk = gn & (DKK - 1);
                v8h ph;
                #pragma unroll
                for (int r = 0; r < 8; ++r) ph[r] = (_Float16)acc[fm][fn][r];
                *(v8h*)&Ch[(((size_t)b * HH + h) * DKK + dk) * TT + tb] = ph;
            } else {
                #pragma unroll
                for (int r = 0; r < 8; ++r)
                    Cg[(size_t)(ml + r) * ldc + gn] = acc[fm][fn][r];
            }
        }
}

// ---------------------------------------------------------------------------
// Fused attention per (head, 64-query strip). Transposed GEMMs put the
// softmax (key) axis in-register; K/V tiles stream via async global->LDS
// copies with double buffering so tile kt+1's HBM traffic overlaps tile kt's
// WMMA + softmax work.
// ---------------------------------------------------------------------------
__global__ __launch_bounds__(128) void attention_fused_kernel(
    const _Float16* __restrict__ Qh,   // [B,H,T,DK]
    const _Float16* __restrict__ Kh,   // [B,H,T,DK]
    const _Float16* __restrict__ Vt,   // [B,H,DK,T]
    float* __restrict__ weights,       // [B,H,T,T]
    float* __restrict__ ctx) {         // [B,T,D]

    constexpr float alpha = 0.125f;    // 1/sqrt(DK)
    constexpr int   NT    = TT / 64;   // 32 key tiles

    const int bh = blockIdx.y;
    const int b  = bh >> 4, h = bh & (HH - 1);
    const _Float16* Qd = Qh + (size_t)bh * TT * DKK;
    const _Float16* Kd = Kh + (size_t)bh * TT * DKK;
    const _Float16* Vd = Vt + (size_t)bh * DKK * TT;
    float* Wd = weights + (size_t)bh * TT * TT;
    float* Cd = ctx + (size_t)b * TT * DD + (size_t)h * DKK;
    const int t0 = blockIdx.x * 64;

    __shared__ alignas(16) _Float16 Qs[64][64];      // [query][dk]
    __shared__ alignas(16) _Float16 Ks[2][64][64];   // [key][dk], double buffered
    __shared__ alignas(16) _Float16 Vs[2][64][64];   // [dk][key], double buffered
    __shared__ alignas(16) _Float16 Ps[64][64];      // [query][key]
    __shared__ float Mrow[64], Srow[64];
    __shared__ float pmax[2][64], psum[2][64];       // [wm][query]

    const int tid  = threadIdx.x;
    const int lane = tid & 31;
    const int wave = tid >> 5;
    const int wm   = wave >> 1, wn = wave & 1;
    const int g    = lane >> 4, lr = lane & 15;

    // Per-thread staging coordinates (4 x 16B per tile buffer).
    const int sn = tid >> 1;                 // 0..63 (row)
    const int ss = (tid & 1) * 4;            // segment base: 0 or 4

    auto issueK = [&](int kt, int buf) {
        const int k0 = kt * 64;
        #pragma unroll
        for (int j = 0; j < 4; ++j)
            async_copy_b128(&Ks[buf][sn][(ss + j) * 8],
                            &Kd[(size_t)(k0 + sn) * DKK + (ss + j) * 8]);
    };
    auto issueKV = [&](int kt, int buf) {
        const int k0 = kt * 64;
        #pragma unroll
        for (int j = 0; j < 4; ++j) {
            async_copy_b128(&Ks[buf][sn][(ss + j) * 8],
                            &Kd[(size_t)(k0 + sn) * DKK + (ss + j) * 8]);
            async_copy_b128(&Vs[buf][sn][(ss + j) * 8],
                            &Vd[(size_t)sn * TT + k0 + (ss + j) * 8]);
        }
    };

    // Stage the Q strip once (64x64 f16), synchronous 16B copies.
    #pragma unroll
    for (int i = 0; i < 4; ++i) {
        int idx = i * 128 + tid;
        int m = idx >> 3, s = idx & 7;
        *(v8h*)&Qs[m][s * 8] = *(const v8h*)&Qd[(size_t)(t0 + m) * DKK + s * 8];
    }
    if (tid < 64) { Mrow[tid] = -3.0e38f; Srow[tid] = 0.0f; }

    // ---------------- Pass A: online softmax statistics ----------------
    issueK(0, 0);
    for (int kt = 0; kt < NT; ++kt) {
        const int cur = kt & 1;
        wait_async0();
        __syncthreads();                       // Ks[cur] ready everywhere
        if (kt + 1 < NT) issueK(kt + 1, cur ^ 1);

        // S^T tile: A = K rows, B = Q rows.
        v8f sacc[2][2] = {};
        #pragma unroll
        for (int kk = 0; kk < 64; kk += 32) {
            v16h af[2], bf[2];
            #pragma unroll
            for (int fm = 0; fm < 2; ++fm) af[fm] = load_a_frag(&Ks[cur][wm*32 + fm*16 + lr][kk], g);
            #pragma unroll
            for (int fn = 0; fn < 2; ++fn) bf[fn] = load_b_frag(&Qs[wn*32 + fn*16 + lr][kk], g);
            #pragma unroll
            for (int fm = 0; fm < 2; ++fm)
                #pragma unroll
                for (int fn = 0; fn < 2; ++fn)
                    sacc[fm][fn] = __builtin_amdgcn_wmma_f32_16x16x32_f16(
                        false, af[fm], false, bf[fn], (short)0, sacc[fm][fn], false, false);
        }

        // Per-query tile max: 16 in-lane values + one xor-16 + LDS combine.
        #pragma unroll
        for (int fn = 0; fn < 2; ++fn) {
            float mv = sacc[0][fn][0];
            #pragma unroll
            for (int fm = 0; fm < 2; ++fm)
                #pragma unroll
                for (int r = 0; r < 8; ++r) mv = fmaxf(mv, sacc[fm][fn][r]);
            mv = fmaxf(mv, __shfl_xor(mv, 16, 32));
            if (g == 0) pmax[wm][wn * 32 + fn * 16 + lr] = mv * alpha;
        }
        __syncthreads();
        if (tid < 64) {
            float mOld = Mrow[tid];
            float mNew = fmaxf(mOld, fmaxf(pmax[0][tid], pmax[1][tid]));
            Srow[tid] *= __expf(mOld - mNew);
            Mrow[tid] = mNew;
        }
        __syncthreads();
        #pragma unroll
        for (int fn = 0; fn < 2; ++fn) {
            int q = wn * 32 + fn * 16 + lr;
            float M = Mrow[q];
            float e = 0.0f;
            #pragma unroll
            for (int fm = 0; fm < 2; ++fm)
                #pragma unroll
                for (int r = 0; r < 8; ++r)
                    e += __expf(alpha * sacc[fm][fn][r] - M);
            e += __shfl_xor(e, 16, 32);
            if (g == 0) psum[wm][q] = e;
        }
        __syncthreads();
        if (tid < 64) Srow[tid] += psum[0][tid] + psum[1][tid];
        __syncthreads();
    }

    if (tid < 64) Srow[tid] = 1.0f / Srow[tid];

    // ---------------- Pass B: weights (single write) + context ----------------
    v8f oacc[2][2] = {};   // ctx^T: M = dk, N = query
    issueKV(0, 0);
    for (int kt = 0; kt < NT; ++kt) {
        const int cur = kt & 1;
        const int k0  = kt * 64;
        wait_async0();
        __syncthreads();                       // Ks/Vs[cur] ready (+ Srow visible)
        if (kt + 1 < NT) issueKV(kt + 1, cur ^ 1);

        v8f sacc[2][2] = {};
        #pragma unroll
        for (int kk = 0; kk < 64; kk += 32) {
            v16h af[2], bf[2];
            #pragma unroll
            for (int fm = 0; fm < 2; ++fm) af[fm] = load_a_frag(&Ks[cur][wm*32 + fm*16 + lr][kk], g);
            #pragma unroll
            for (int fn = 0; fn < 2; ++fn) bf[fn] = load_b_frag(&Qs[wn*32 + fn*16 + lr][kk], g);
            #pragma unroll
            for (int fm = 0; fm < 2; ++fm)
                #pragma unroll
                for (int fn = 0; fn < 2; ++fn)
                    sacc[fm][fn] = __builtin_amdgcn_wmma_f32_16x16x32_f16(
                        false, af[fm], false, bf[fn], (short)0, sacc[fm][fn], false, false);
        }

        // Normalize; per fragment: 8 consecutive keys -> packed stores.
        #pragma unroll
        for (int fn = 0; fn < 2; ++fn) {
            int q = wn * 32 + fn * 16 + lr;
            float M  = Mrow[q];
            float is = Srow[q];
            float* wrow = Wd + (size_t)(t0 + q) * TT + k0;
            #pragma unroll
            for (int fm = 0; fm < 2; ++fm) {
                int kb = wm * 32 + fm * 16 + g * 8;     // key base (8 consecutive)
                v4f w0, w1;
                v8h ph;
                #pragma unroll
                for (int r = 0; r < 8; ++r) {
                    float p = __expf(alpha * sacc[fm][fn][r] - M) * is;
                    if (r < 4) w0[r] = p; else w1[r - 4] = p;
                    ph[r] = (_Float16)p;
                }
                *(v4f*)(wrow + kb)     = w0;
                *(v4f*)(wrow + kb + 4) = w1;
                *(v8h*)&Ps[q][kb]      = ph;
            }
        }
        __syncthreads();

        // ctx^T accumulate: A = Vs rows (dk-major), B = Ps rows.
        #pragma unroll
        for (int kk = 0; kk < 64; kk += 32) {
            v16h af[2], bf[2];
            #pragma unroll
            for (int fm = 0; fm < 2; ++fm) af[fm] = load_a_frag(&Vs[cur][wm*32 + fm*16 + lr][kk], g);
            #pragma unroll
            for (int fn = 0; fn < 2; ++fn) bf[fn] = load_b_frag(&Ps[wn*32 + fn*16 + lr][kk], g);
            #pragma unroll
            for (int fm = 0; fm < 2; ++fm)
                #pragma unroll
                for (int fn = 0; fn < 2; ++fn)
                    oacc[fm][fn] = __builtin_amdgcn_wmma_f32_16x16x32_f16(
                        false, af[fm], false, bf[fn], (short)0, oacc[fm][fn], false, false);
        }
        __syncthreads();
    }

    // Write context tile: per fragment, 8 consecutive dk -> packed f32 stores.
    #pragma unroll
    for (int fn = 0; fn < 2; ++fn) {
        int q = wn * 32 + fn * 16 + lr;
        float* crow = Cd + (size_t)(t0 + q) * DD;
        #pragma unroll
        for (int fm = 0; fm < 2; ++fm) {
            int dkb = wm * 32 + fm * 16 + g * 8;
            v4f c0, c1;
            #pragma unroll
            for (int r = 0; r < 8; ++r) {
                float v = oacc[fm][fn][r];
                if (r < 4) c0[r] = v; else c1[r - 4] = v;
            }
            *(v4f*)(crow + dkb)     = c0;
            *(v4f*)(crow + dkb + 4) = c1;
        }
    }
}

// ---------------------------------------------------------------------------
// Host-side orchestration.
// ---------------------------------------------------------------------------
extern "C" void kernel_launch(void* const* d_in, const int* in_sizes, int n_in,
                              void* d_out, int out_size, void* d_ws, size_t ws_size,
                              hipStream_t stream) {
    (void)in_sizes; (void)n_in; (void)out_size; (void)ws_size;

    const float* query = (const float*)d_in[0];
    const float* key_  = (const float*)d_in[1];
    const float* value = (const float*)d_in[2];
    const float* W_q   = (const float*)d_in[3];
    const float* W_k   = (const float*)d_in[4];
    const float* W_v   = (const float*)d_in[5];
    const float* W_o   = (const float*)d_in[6];

    const size_t qkvElems = (size_t)BB * HH * TT * DKK;  // 4,194,304
    _Float16* Qh = (_Float16*)d_ws;
    _Float16* Kh = Qh + qkvElems;
    _Float16* Vt = Kh + qkvElems;                 // dk-major per head
    float*    ctx = (float*)(Vt + qkvElems);      // B*T*D f32

    float* out_proj = (float*)d_out;                    // (B,T,D)
    float* weights  = out_proj + (size_t)BB * TT * DD;  // (B,H,T,T)

    const dim3 blk(128);

    // 1) Projections. Q/K: transposed orientation (grid = tokens x features).
    {
        dim3 gridT((BB * TT) / 64, DD / 64);
        proj_gemm_kernel<1><<<gridT, blk, 0, stream>>>(query, W_q, nullptr, Qh, DD, DD, DD, 0);
        proj_gemm_kernel<1><<<gridT, blk, 0, stream>>>(key_, W_k, nullptr, Kh, DD, DD, DD, 0);
        dim3 gridN(DD / 64, (BB * TT) / 64);
        proj_gemm_kernel<3><<<gridN, blk, 0, stream>>>(value, W_v, nullptr, Vt, DD, DD, DD, 0);
    }

    // 2) Fused attention: stats -> weights (single write) + context.
    attention_fused_kernel<<<dim3(TT / 64, BB * HH), blk, 0, stream>>>(
        Qh, Kh, Vt, weights, ctx);

    // 3) Output projection (normal orientation, f32 out).
    {
        dim3 gridN(DD / 64, (BB * TT) / 64);
        proj_gemm_kernel<0><<<gridN, blk, 0, stream>>>(ctx, W_o, out_proj, nullptr, DD, DD, DD, DD);
    }
}
